// GroupedQueryAttention_36421322670778
// MI455X (gfx1250) — matrix operand order repaired and checked by reading, not compiled, not run on hardware
//
#include <hip/hip_runtime.h>
#include <stdint.h>

// ---------- types ----------
typedef __attribute__((ext_vector_type(16))) __bf16 v16bf;
typedef __attribute__((ext_vector_type(8)))  float  v8f;
typedef __attribute__((ext_vector_type(4)))  unsigned int v4u;
typedef __attribute__((ext_vector_type(8)))  int    v8i;
typedef __attribute__((ext_vector_type(4)))  int    v4i;

__device__ __forceinline__ unsigned short f32_to_bf16(float f) {
  unsigned int u = __float_as_uint(f);
  u += 0x7FFFu + ((u >> 16) & 1u);           // round-to-nearest-even
  return (unsigned short)(u >> 16);
}
__device__ __forceinline__ float bf16_to_f32(unsigned short h) {
  return __uint_as_float(((unsigned int)h) << 16);
}

struct B32 { uint4 a, b; };
__device__ __forceinline__ v16bf frag16(const unsigned short* p0, const unsigned short* p1) {
  B32 u;
  u.a = *(const uint4*)p0;
  u.b = *(const uint4*)p1;
  return __builtin_bit_cast(v16bf, u);
}
struct PW8 { unsigned int w[8]; };

// ---------- LDS byte offset of a __shared__ object (for TDM descriptors) ----------
__device__ __forceinline__ unsigned int lds_offset(const void* p) {
  return (unsigned int)(unsigned long long)(const __attribute__((address_space(3))) void*)p;
}

// ---------- Tensor Data Mover: 2D tile load Global -> LDS ----------
// Tile: tile_h rows x tile_w elements (2-byte), row stride = row_stride elements.
// Rows are packed contiguously into LDS at lds_byte (matches [rows][tile_w] layout).
__device__ __forceinline__ void tdm_load_2d(const unsigned short* gsrc, unsigned int lds_byte,
                                            int tile_w, int tile_h, int row_stride) {
  unsigned long long ga = (unsigned long long)(uintptr_t)gsrc;
  v4u g0;
  g0[0] = 1u;                                           // count=1, user descriptor
  g0[1] = lds_byte;                                     // lds_addr (bytes)
  g0[2] = (unsigned int)ga;                             // global_addr[31:0]
  g0[3] = (unsigned int)((ga >> 32) & 0x1FFFFFFull)     // global_addr[56:32]
        | (2u << 30);                                   // type = 2 ("image")
  v8i g1;
  g1[0] = 1 << 16;                                      // data_size = 1 (2 bytes)
  g1[1] = (int)((unsigned)(row_stride & 0xFFFF) << 16); // tensor_dim0[15:0]
  g1[2] = (int)(((unsigned)row_stride >> 16) & 0xFFFFu);// tensor_dim0[31:16], dim1 lo=big
  g1[3] = (int)(0x10u | ((unsigned)tile_w << 16));      // tensor_dim1 hi (=1M rows), tile_dim0
  g1[4] = tile_h;                                       // tile_dim1 (tile_dim2 = 0)
  g1[5] = row_stride;                                   // tensor_dim0_stride lo32 (elements)
  g1[6] = 0;                                            // stride0 hi / stride1 lo
  g1[7] = 0;                                            // stride1 hi
  v4i z4 = {};
  v8i z8 = {};
  __builtin_amdgcn_tensor_load_to_lds(g0, g1, z4, z4, z8, 0);
}

// ---------- f32 -> bf16 convert ----------
__global__ __launch_bounds__(256) void cvt_f32_bf16(const float* __restrict__ in,
                                                    unsigned short* __restrict__ out, int n4) {
  int i = blockIdx.x * 256 + threadIdx.x;
  if (i < n4) {
    float4 v = *(const float4*)(in + (size_t)i * 4);
    uint2 w;
    w.x = (unsigned int)f32_to_bf16(v.x) | ((unsigned int)f32_to_bf16(v.y) << 16);
    w.y = (unsigned int)f32_to_bf16(v.z) | ((unsigned int)f32_to_bf16(v.w) << 16);
    *(uint2*)(out + (size_t)i * 4) = w;
  }
}

// ---------- GEMM: C[M,N] = A[M,K] * B[N,K]^T, bf16 in, f32 acc ----------
// Block tile 128x128, K-step 32. TDM (tensor_load_to_lds) double-buffers the
// A/B LDS slabs while 8 waves (4x2) compute 32x64 tiles (8 WMMAs each).
template <bool OUT_BF16>
__global__ __launch_bounds__(256) void gemm_bf16(const unsigned short* __restrict__ A,
                                                 const unsigned short* __restrict__ B,
                                                 void* __restrict__ C,
                                                 int M, int N, int K) {
  __shared__ unsigned short As[2][128 * 32];
  __shared__ unsigned short Bs[2][128 * 32];

  const int tid  = threadIdx.x;
  const int lane = tid & 31;
  const int wave = tid >> 5;
  const int wm   = (wave & 3) * 32;           // 0..96
  const int wn   = (wave >> 2) * 64;          // 0 / 64
  const int bm   = blockIdx.y * 128;
  const int bn   = blockIdx.x * 128;

  const int mrow = lane & 15;
  const int koff = (lane >> 4) << 3;          // A-frag half-wave chunk: 0 / 8
  const int boff = (lane >> 4) << 4;          // half-wave chunk 0 / 16 (unused; B-frags use koff)

  v8f zero = {};
  v8f acc[2][4];
#pragma unroll
  for (int i = 0; i < 2; ++i)
#pragma unroll
    for (int j = 0; j < 4; ++j) acc[i][j] = zero;

  const unsigned short* Abase = A + (size_t)bm * K;
  const unsigned short* Bbase = B + (size_t)bn * K;

  // prologue: TDM-load the first K-slab
  if (wave == 0) {
    tdm_load_2d(Abase, lds_offset(&As[0][0]), 32, 128, K);
    tdm_load_2d(Bbase, lds_offset(&Bs[0][0]), 32, 128, K);
    __builtin_amdgcn_s_wait_tensorcnt(0);
  }
  __syncthreads();

  int cur = 0;
  for (int k0 = 0; k0 < K; k0 += 32) {
    if (k0 + 32 < K && wave == 0) {           // kick off next slab; overlaps WMMA below
      tdm_load_2d(Abase + k0 + 32, lds_offset(&As[cur ^ 1][0]), 32, 128, K);
      tdm_load_2d(Bbase + k0 + 32, lds_offset(&Bs[cur ^ 1][0]), 32, 128, K);
    }

    v16bf af[2], bf[4];
#pragma unroll
    for (int i = 0; i < 2; ++i)
      af[i] = frag16(&As[cur][(wm + i * 16 + mrow) * 32 + koff],
                     &As[cur][(wm + i * 16 + mrow) * 32 + koff + 16]);
#pragma unroll
    for (int j = 0; j < 4; ++j)
      bf[j] = frag16(&Bs[cur][(wn + j * 16 + mrow) * 32 + koff],
                     &Bs[cur][(wn + j * 16 + mrow) * 32 + koff + 16]);
#pragma unroll
    for (int i = 0; i < 2; ++i)
#pragma unroll
      for (int j = 0; j < 4; ++j)
        acc[i][j] = __builtin_amdgcn_wmma_f32_16x16x32_bf16(false, af[i], false, bf[j],
                                                            (short)0, acc[i][j], false, false);

    if (wave == 0) __builtin_amdgcn_s_wait_tensorcnt(0);
    __syncthreads();
    cur ^= 1;
  }

  // C/D layout: lane holds column n=lane&15; rows = vgpr + 8*(lane>=16)
  const int ncol = lane & 15;
#pragma unroll
  for (int i = 0; i < 2; ++i)
#pragma unroll
    for (int j = 0; j < 4; ++j)
#pragma unroll
      for (int r = 0; r < 8; ++r) {
        int m = bm + wm + i * 16 + koff + r;
        int n = bn + wn + j * 16 + ncol;
        if (OUT_BF16) ((unsigned short*)C)[(size_t)m * N + n] = f32_to_bf16(acc[i][j][r]);
        else          ((float*)C)[(size_t)m * N + n] = acc[i][j][r];
      }
}

// ---------- RoPE (in place on bf16 qkv), pre-scales Q by 1/sqrt(128) ----------
__global__ __launch_bounds__(256) void rope_kernel(unsigned short* __restrict__ qkv, int S) {
  const int tid  = threadIdx.x;
  const int d    = tid & 63;
  const int hsub = tid >> 6;                 // 0..3
  const int grp  = blockIdx.x % 10;
  const int bs   = blockIdx.x / 10;          // b*S + s
  const int s    = bs % S;
  const int head = grp * 4 + hsub;           // 0..39 (32 q heads + 8 k heads)
  const int feat = (head < 32) ? head * 128 : 4096 + (head - 32) * 128;
  const size_t base = (size_t)bs * 6144 + feat;

  const float inv_freq = __expf(-(float)d * 0.14391156831212787f);  // ln(1e4)/64
  float c, sn;
  __sincosf((float)s * inv_freq, &sn, &c);
  float x1 = bf16_to_f32(qkv[base + d]);
  float x2 = bf16_to_f32(qkv[base + 64 + d]);
  float o1 = x1 * c - x2 * sn;
  float o2 = x2 * c + x1 * sn;
  if (head < 32) { o1 *= 0.08838834764831845f; o2 *= 0.08838834764831845f; }
  qkv[base + d]      = f32_to_bf16(o1);
  qkv[base + 64 + d] = f32_to_bf16(o2);
}

// ---------- Flash attention (GQA): 4 waves = 4 grouped Q heads share K/V LDS ----------
__global__ __launch_bounds__(128) void attn_kernel(const unsigned short* __restrict__ qkv,
                                                   unsigned short* __restrict__ attn, int S) {
  __shared__ unsigned short Ks[32 * 128];    // K block: 32 keys x 128d
  __shared__ unsigned short Vt[128 * 40];    // V^T block: 128d x 32 keys (padded stride 40)

  const int QT = S >> 4;
  const int qt = blockIdx.x % QT;
  const int hk = (blockIdx.x / QT) & 7;
  const int b  = blockIdx.x / (QT * 8);

  const int tid  = threadIdx.x;
  const int lane = tid & 31;
  const int wave = tid >> 5;                 // group member 0..3
  const int h    = hk * 4 + wave;            // q head

  const int n    = lane & 15;
  const int koff = (lane >> 4) << 3;         // A-frag chunk 0/8
  const int boff = (lane >> 4) << 4;         // V^T A-frag chunk 0/16

  // Q fragments in B-operand layout (Q^T columns = q rows)
  v16bf qf[4];
  {
    const size_t qb = ((size_t)b * S + (size_t)qt * 16 + n) * 6144 + (size_t)h * 128;
#pragma unroll
    for (int c = 0; c < 4; ++c) {
      int d0 = c * 32 + koff;
      qf[c] = frag16(&qkv[qb + d0], &qkv[qb + d0 + 16]);
    }
  }

  v8f zero = {};
  v8f o[8];
#pragma unroll
  for (int c = 0; c < 8; ++c) o[c] = zero;

  float mrun = -1e30f, lrun = 0.0f;
  const int qi  = qt * 16 + n;
  const int nkb = (qt * 16 + 47) >> 5;       // # of 32-key blocks to cover causal range

  for (int kb = 0; kb < nkb * 32; kb += 32) {
    __syncthreads();
    // stage K (row major) and V^T (transpose scatter) in LDS; 128 threads, 2 chunks each
#pragma unroll
    for (int it = 0; it < 2; ++it) {
      int cc  = tid + it * 128;
      int row = cc >> 3;                     // key 0..31
      int col = (cc & 7) << 3;               // d chunk
      const size_t kr = ((size_t)b * S + kb + row) * 6144;
      *(uint4*)&Ks[row * 128 + col] = *(const uint4*)&qkv[kr + 4096 + hk * 128 + col];
      uint4 v = *(const uint4*)&qkv[kr + 5120 + hk * 128 + col];
      Vt[(col + 0) * 40 + row] = (unsigned short)(v.x);
      Vt[(col + 1) * 40 + row] = (unsigned short)(v.x >> 16);
      Vt[(col + 2) * 40 + row] = (unsigned short)(v.y);
      Vt[(col + 3) * 40 + row] = (unsigned short)(v.y >> 16);
      Vt[(col + 4) * 40 + row] = (unsigned short)(v.z);
      Vt[(col + 5) * 40 + row] = (unsigned short)(v.z >> 16);
      Vt[(col + 6) * 40 + row] = (unsigned short)(v.w);
      Vt[(col + 7) * 40 + row] = (unsigned short)(v.w >> 16);
    }
    __syncthreads();

    // S' = K x Q^T  (transposed scores: lane column = q row)
    v8f s0 = zero, s1 = zero;
#pragma unroll
    for (int c = 0; c < 4; ++c) {
      int d0 = c * 32 + koff;
      v16bf k0 = frag16(&Ks[n * 128 + d0],        &Ks[n * 128 + d0 + 16]);
      v16bf k1 = frag16(&Ks[(16 + n) * 128 + d0], &Ks[(16 + n) * 128 + d0 + 16]);
      s0 = __builtin_amdgcn_wmma_f32_16x16x32_bf16(false, k0, false, qf[c], (short)0, s0, false, false);
      s1 = __builtin_amdgcn_wmma_f32_16x16x32_bf16(false, k1, false, qf[c], (short)0, s1, false, false);
    }

    // causal mask + online softmax (per-lane stats; xor-16 merges key halves)
    float sv0[8], sv1[8], bmax = -1e30f;
#pragma unroll
    for (int r = 0; r < 8; ++r) {
      int key0 = kb + koff + r;
      int key1 = kb + 16 + koff + r;
      sv0[r] = (key0 <= qi) ? s0[r] : -1e30f;
      sv1[r] = (key1 <= qi) ? s1[r] : -1e30f;
      bmax = fmaxf(bmax, fmaxf(sv0[r], sv1[r]));
    }
    bmax = fmaxf(bmax, __shfl_xor(bmax, 16));
    float mnew  = fmaxf(mrun, bmax);
    float alpha = __expf(mrun - mnew);
    float psum  = 0.0f;
    unsigned int pw0[4], pw1[4];
#pragma unroll
    for (int r = 0; r < 4; ++r) {
      float p0 = __expf(sv0[2 * r] - mnew);
      float p1 = __expf(sv0[2 * r + 1] - mnew);
      psum += p0 + p1;
      pw0[r] = (unsigned int)f32_to_bf16(p0) | ((unsigned int)f32_to_bf16(p1) << 16);
    }
#pragma unroll
    for (int r = 0; r < 4; ++r) {
      float p0 = __expf(sv1[2 * r] - mnew);
      float p1 = __expf(sv1[2 * r + 1] - mnew);
      psum += p0 + p1;
      pw1[r] = (unsigned int)f32_to_bf16(p0) | ((unsigned int)f32_to_bf16(p1) << 16);
    }
    psum += __shfl_xor(psum, 16);
    lrun = lrun * alpha + psum;
    mrun = mnew;

    // P' into B-operand layout: lanes<16 need keys 0..15, lanes>=16 need 16..31
    PW8 pws;
#pragma unroll
    for (int r = 0; r < 4; ++r) {
      unsigned int x0 = (unsigned int)__shfl_xor((int)pw0[r], 16);
      unsigned int x1 = (unsigned int)__shfl_xor((int)pw1[r], 16);
      pws.w[r]     = (lane < 16) ? pw0[r] : x1;
      pws.w[4 + r] = (lane < 16) ? x0 : pw1[r];
    }
    v16bf pf = __builtin_bit_cast(v16bf, pws);

    // O' += V^T x P'  (8 d-chunks of 16)
#pragma unroll
    for (int c = 0; c < 8; ++c) {
#pragma unroll
      for (int r = 0; r < 8; ++r) o[c][r] *= alpha;
      v16bf vf = frag16(&Vt[(c * 16 + n) * 40 + boff], &Vt[(c * 16 + n) * 40 + boff + 8]);
      o[c] = __builtin_amdgcn_wmma_f32_16x16x32_bf16(false, vf, false, pf, (short)0, o[c], false, false);
    }
  }

  // normalize + store bf16: lane column = q row n; d = c*16 + koff + r
  const float inv = 1.0f / lrun;
  const size_t ob = ((size_t)b * S + (size_t)qt * 16 + n) * 4096 + (size_t)h * 128;
#pragma unroll
  for (int c = 0; c < 8; ++c)
#pragma unroll
    for (int r = 0; r < 4; ++r) {
      unsigned int w = (unsigned int)f32_to_bf16(o[c][2 * r] * inv) |
                       ((unsigned int)f32_to_bf16(o[c][2 * r + 1] * inv) << 16);
      *(unsigned int*)&attn[ob + c * 16 + koff + 2 * r] = w;
    }
}

// ---------- launch ----------
extern "C" void kernel_launch(void* const* d_in, const int* in_sizes, int n_in,
                              void* d_out, int out_size, void* d_ws, size_t ws_size,
                              hipStream_t stream) {
  const float* x     = (const float*)d_in[0];
  const float* w_in  = (const float*)d_in[1];
  const float* w_out = (const float*)d_in[2];

  const int B = 2, S = 2048, E = 4096, F = 6144;
  const int M = B * S;  // 4096

  char* ws = (char*)d_ws;
  size_t off = 0;
  unsigned short* xb   = (unsigned short*)(ws + off); off += 2ull * M * E;   // 32 MB
  unsigned short* wib  = (unsigned short*)(ws + off); off += 2ull * F * E;   // 48 MB
  unsigned short* wob  = (unsigned short*)(ws + off); off += 2ull * E * E;   // 32 MB
  unsigned short* qkvb = (unsigned short*)(ws + off); off += 2ull * M * F;   // 48 MB
  unsigned short* attb = (unsigned short*)(ws + off); off += 2ull * M * E;   // 32 MB

  cvt_f32_bf16<<<(M * E / 4) / 256, 256, 0, stream>>>(x, xb, M * E / 4);
  cvt_f32_bf16<<<(F * E / 4) / 256, 256, 0, stream>>>(w_in, wib, F * E / 4);
  cvt_f32_bf16<<<(E * E / 4) / 256, 256, 0, stream>>>(w_out, wob, E * E / 4);

  gemm_bf16<true><<<dim3(F / 128, M / 128), 256, 0, stream>>>(xb, wib, qkvb, M, F, E);
  rope_kernel<<<B * S * 10, 256, 0, stream>>>(qkvb, S);
  attn_kernel<<<B * 8 * (S / 16), 128, 0, stream>>>(qkvb, attb, S);
  gemm_bf16<false><<<dim3(E / 128, M / 128), 256, 0, stream>>>(attb, wob, (float*)d_out, M, E, E);
}
